// PGWCrossAttention_31817117728902
// MI455X (gfx1250) — compile-verified, matrix operand order checked by reading, not run on hardware
//
#include <hip/hip_runtime.h>

typedef __attribute__((ext_vector_type(16))) _Float16 v16h;
typedef __attribute__((ext_vector_type(8)))  float    v8f;

#define BB    4
#define LQ    1024
#define LK    2048
#define HH    8
#define DD    32
#define FF    11
#define SPN   8
#define WDN   3
#define HIDN  256
#define TEMPR 0.1f
#define LOG2E 1.44269504088896340736f

// ---------------------------------------------------------------------------
// helpers
// ---------------------------------------------------------------------------
__device__ __forceinline__ v8f wmma16x16x32(v16h a, v16h b, v8f c) {
  // D = A(16x32 f16) * B(32x16 f16) + C(16x16 f32)
  return __builtin_amdgcn_wmma_f32_16x16x32_f16(false, a, false, b, (short)0, c,
                                                false, false);
}

__device__ __forceinline__ float sigmoidf_(float x) {
  return 1.0f / (1.0f + __expf(-x));
}

// half element e of a 16-half WMMA operand (A or B, f16) maps to inner index
// within a 32-wide K/D chunk:  ((e&8)?16:0) + 8*khalf + (e&7),  khalf = lane>>4
__device__ __forceinline__ int elem_off(int e, int khalf) {
  return ((e & 8) ? 16 : 0) + khalf * 8 + (e & 7);
}

__device__ __forceinline__ v16h cvt16(float4 a, float4 b, float4 c, float4 d) {
  v16h r;
  r[0]  = (_Float16)a.x; r[1]  = (_Float16)a.y; r[2]  = (_Float16)a.z; r[3]  = (_Float16)a.w;
  r[4]  = (_Float16)b.x; r[5]  = (_Float16)b.y; r[6]  = (_Float16)b.z; r[7]  = (_Float16)b.w;
  r[8]  = (_Float16)c.x; r[9]  = (_Float16)c.y; r[10] = (_Float16)c.z; r[11] = (_Float16)c.w;
  r[12] = (_Float16)d.x; r[13] = (_Float16)d.y; r[14] = (_Float16)d.z; r[15] = (_Float16)d.w;
  return r;
}

// ---------------------------------------------------------------------------
// log2(pi) precompute via hardware v_log_f32 (shared across 8 heads).
// All logits live in the log2 domain so softmax uses native exp2.
// ---------------------------------------------------------------------------
__global__ void logpi_kernel(const float* __restrict__ pi, float* __restrict__ lp) {
  int t = blockIdx.x * blockDim.x + threadIdx.x;
  lp[t] = __builtin_amdgcn_logf(fmaxf(pi[t], 1e-9f));
}

// ---------------------------------------------------------------------------
// Q' pack: concat(c1*q_struct, c2*q_wp) per head (c1,c2 carry log2e),
// stored in the per-lane f16 B-operand layout:
// [bh][qtile][dchunk(2)][lane(32)][16 halves].  Thread id == flat slot index.
// ---------------------------------------------------------------------------
__global__ void pack_q_kernel(const float* __restrict__ q_fp,
                              const float* __restrict__ Wq_s, const float* __restrict__ bq_s,
                              const float* __restrict__ Wq_w, const float* __restrict__ bq_w,
                              const float* __restrict__ wave_blend,
                              _Float16* __restrict__ Qp) {
  int t    = blockIdx.x * blockDim.x + threadIdx.x;   // < 4*8*64*2*32 = 131072
  int lane = t & 31;
  int dt   = (t >> 5) & 1;       // 0: struct dims 0..31, 1: wave dims 32..63
  int qt   = (t >> 6) & 63;
  int bh   = t >> 12;
  int b = bh >> 3, h = bh & 7;
  int q = qt * 16 + (lane & 15);
  int khalf = lane >> 4;

  float alpha = sigmoidf_(wave_blend[0]);
  float c1 = LOG2E * (1.0f - alpha) / sqrtf((float)DD);
  float c2 = LOG2E * 2.0f * alpha / TEMPR;

  const float* x = q_fp + (size_t)(b * LQ + q) * FF;
  float xv[FF];
#pragma unroll
  for (int f = 0; f < FF; ++f) xv[f] = x[f];

  v16h ov;
  if (dt == 0) {
#pragma unroll
    for (int e = 0; e < 16; ++e) {
      int c = h * DD + elem_off(e, khalf);
      const float* w = Wq_s + c * SPN;
      float v = bq_s[c];
#pragma unroll
      for (int s = 0; s < SPN; ++s) v = fmaf(xv[s], w[s], v);
      ov[e] = (_Float16)(v * c1);
    }
  } else {
#pragma unroll
    for (int e = 0; e < 16; ++e) {
      int c = h * DD + elem_off(e, khalf);
      const float* w = Wq_w + c * WDN;
      float v = bq_w[c];
#pragma unroll
      for (int s = 0; s < WDN; ++s) v = fmaf(xv[SPN + s], w[s], v);
      ov[e] = (_Float16)(v * c2);
    }
  }
  *reinterpret_cast<v16h*>(Qp + (size_t)t * 16) = ov;
}

// ---------------------------------------------------------------------------
// K' pack: concat(k_struct, k_wp) (unscaled) in A-operand layout:
// [bh][ktile(128)][dchunk(2)][lane][16 halves]
// ---------------------------------------------------------------------------
__global__ void pack_k_kernel(const float* __restrict__ v_ret,
                              const float* __restrict__ Wk_s, const float* __restrict__ bk_s,
                              const float* __restrict__ Wk_w, const float* __restrict__ bk_w,
                              _Float16* __restrict__ Kp) {
  int t    = blockIdx.x * blockDim.x + threadIdx.x;   // < 4*8*128*2*32 = 262144
  int lane = t & 31;
  int dh   = (t >> 5) & 1;
  int kt   = (t >> 6) & 127;
  int bh   = t >> 13;
  int b = bh >> 3, h = bh & 7;
  int k = kt * 16 + (lane & 15);
  int khalf = lane >> 4;

  const float* x = v_ret + (size_t)(b * LK + k) * FF;
  float xv[FF];
#pragma unroll
  for (int f = 0; f < FF; ++f) xv[f] = x[f];

  v16h ov;
  if (dh == 0) {
#pragma unroll
    for (int e = 0; e < 16; ++e) {
      int c = h * DD + elem_off(e, khalf);
      const float* w = Wk_s + c * SPN;
      float v = bk_s[c];
#pragma unroll
      for (int s = 0; s < SPN; ++s) v = fmaf(xv[s], w[s], v);
      ov[e] = (_Float16)v;
    }
  } else {
#pragma unroll
    for (int e = 0; e < 16; ++e) {
      int c = h * DD + elem_off(e, khalf);
      const float* w = Wk_w + c * WDN;
      float v = bk_w[c];
#pragma unroll
      for (int s = 0; s < WDN; ++s) v = fmaf(xv[SPN + s], w[s], v);
      ov[e] = (_Float16)v;
    }
  }
  *reinterpret_cast<v16h*>(Kp + (size_t)t * 16) = ov;
}

// ---------------------------------------------------------------------------
// V^T pack (for ctx^T = V^T * P^T), A-operand layout over k-chunks of 32:
// [bh][kchunk(64)][dtile(2)][lane][16 halves];   row m = d, K = k
// ---------------------------------------------------------------------------
__global__ void pack_v_kernel(const float* __restrict__ v_ret,
                              const float* __restrict__ Wv, const float* __restrict__ bv,
                              _Float16* __restrict__ Vp) {
  int t    = blockIdx.x * blockDim.x + threadIdx.x;   // < 4*8*64*2*32 = 131072
  int lane = t & 31;
  int dt   = (t >> 5) & 1;
  int kc   = (t >> 6) & 63;
  int bh   = t >> 12;
  int b = bh >> 3, h = bh & 7;
  int khalf = lane >> 4;
  int d = dt * 16 + (lane & 15);
  int c = h * DD + d;

  float w[FF];
#pragma unroll
  for (int f = 0; f < FF; ++f) w[f] = Wv[c * FF + f];
  float bias = bv[c];

  v16h ov;
#pragma unroll
  for (int e = 0; e < 16; ++e) {
    int k = kc * 32 + elem_off(e, khalf);
    const float* vr = v_ret + (size_t)(b * LK + k) * FF;
    float v = bias;
#pragma unroll
    for (int f = 0; f < FF; ++f) v = fmaf(vr[f], w[f], v);
    ov[e] = (_Float16)v;
  }
  *reinterpret_cast<v16h*>(Vp + (size_t)t * 16) = ov;
}

// ---------------------------------------------------------------------------
// kbias[b,h,k] = -log2e*(alpha/TEMP)*||k_wp||^2   (||q_wp||^2 term is
// row-constant along k and cancels in softmax)
// ---------------------------------------------------------------------------
__global__ void kbias_kernel(const float* __restrict__ v_ret,
                             const float* __restrict__ Wk_w, const float* __restrict__ bk_w,
                             const float* __restrict__ wave_blend,
                             float* __restrict__ kbias) {
  int t = blockIdx.x * blockDim.x + threadIdx.x;      // < 4*8*2048 = 65536
  int k = t & (LK - 1);
  int h = (t >> 11) & 7;
  int b = t >> 14;
  const float* x = v_ret + (size_t)(b * LK + k) * FF + SPN;
  float x0 = x[0], x1 = x[1], x2 = x[2];
  float acc = 0.0f;
#pragma unroll
  for (int d = 0; d < DD; ++d) {
    int c = h * DD + d;
    const float* w = Wk_w + c * WDN;
    float y = bk_w[c];
    y = fmaf(x0, w[0], y);
    y = fmaf(x1, w[1], y);
    y = fmaf(x2, w[2], y);
    acc = fmaf(y, y, acc);
  }
  float alpha = sigmoidf_(wave_blend[0]);
  kbias[t] = -(LOG2E * alpha / TEMPR) * acc;
}

// ---------------------------------------------------------------------------
// Online-softmax tile update (log2 domain, native exp2). Returns the f16
// P^T operand for the PV WMMA; the f32 C layout of S^T converts to the
// f16 B-operand layout entirely in-lane (zero cross-lane traffic).
// ---------------------------------------------------------------------------
__device__ __forceinline__ v16h flash_update(v8f c0, v8f c1,
                                             const float* __restrict__ lp_row, int kb,
                                             float4 G0, float4 G1, float4 G2, float4 G3,
                                             float& m, float& lsum,
                                             v8f& o0, v8f& o1) {
  float4 L0 = *reinterpret_cast<const float4*>(lp_row + kb);
  float4 L1 = *reinterpret_cast<const float4*>(lp_row + kb + 4);
  float4 L2 = *reinterpret_cast<const float4*>(lp_row + kb + 16);
  float4 L3 = *reinterpret_cast<const float4*>(lp_row + kb + 20);

  float s[16];
  s[0]  = c0[0] + L0.x + G0.x;  s[1]  = c0[1] + L0.y + G0.y;
  s[2]  = c0[2] + L0.z + G0.z;  s[3]  = c0[3] + L0.w + G0.w;
  s[4]  = c0[4] + L1.x + G1.x;  s[5]  = c0[5] + L1.y + G1.y;
  s[6]  = c0[6] + L1.z + G1.z;  s[7]  = c0[7] + L1.w + G1.w;
  s[8]  = c1[0] + L2.x + G2.x;  s[9]  = c1[1] + L2.y + G2.y;
  s[10] = c1[2] + L2.z + G2.z;  s[11] = c1[3] + L2.w + G2.w;
  s[12] = c1[4] + L3.x + G3.x;  s[13] = c1[5] + L3.y + G3.y;
  s[14] = c1[6] + L3.z + G3.z;  s[15] = c1[7] + L3.w + G3.w;

  float cm = s[0];
#pragma unroll
  for (int i = 1; i < 16; ++i) cm = fmaxf(cm, s[i]);
  cm = fmaxf(cm, __shfl_xor(cm, 16, 32));        // combine with partner half-rows
  float mnew = fmaxf(m, cm);
  float scale = __builtin_amdgcn_exp2f(m - mnew);

  float p[16]; float psum = 0.0f;
#pragma unroll
  for (int i = 0; i < 16; ++i) { p[i] = __builtin_amdgcn_exp2f(s[i] - mnew); psum += p[i]; }
  psum += __shfl_xor(psum, 16, 32);
  lsum = lsum * scale + psum;
  m = mnew;

#pragma unroll
  for (int r = 0; r < 8; ++r) { o0[r] *= scale; o1[r] *= scale; }

  v16h pb;
#pragma unroll
  for (int e = 0; e < 16; ++e) pb[e] = (_Float16)p[e];
  return pb;
}

// ---------------------------------------------------------------------------
// Fused flash attention. One wave32 per (b, h, 16-row q-tile): per 32-k step
// 4 WMMAs build S^T (folded spatial+wave logits, D=64), the online softmax
// runs per-lane + two shfl_xor(16), and 2 WMMAs accumulate ctx^T.
// Single-q-tile: fits registers with zero spills (2-tile variant spilled).
// ---------------------------------------------------------------------------
__global__ void flash_kernel(const _Float16* __restrict__ Qp,
                             const _Float16* __restrict__ Kp,
                             const _Float16* __restrict__ Vp,
                             const float* __restrict__ kbias,
                             const float* __restrict__ logpi,
                             float* __restrict__ ctx) {
  int lane = threadIdx.x & 31;
  int wave = threadIdx.x >> 5;
  int tile = blockIdx.x * 4 + wave;        // exactly 2048 tiles
  int qt = tile & 63;
  int bh = tile >> 6;
  int b = bh >> 3, h = bh & 7;
  int khalf = lane >> 4, qcol = lane & 15;
  int q = qt * 16 + qcol;

  const v16h* Qt = reinterpret_cast<const v16h*>(Qp) + (size_t)(bh * 64 + qt) * 64;
  v16h qb0 = Qt[lane];         // Q' dims 0..31  (B operand)
  v16h qb1 = Qt[32 + lane];    // Q' dims 32..63

  const float* lp_row = logpi + (size_t)(b * LQ + q) * LK;
  const float* kb_row = kbias + (size_t)bh * LK;

  v8f o0 = {};                 // ctx^T rows d=0..15  (C layout)
  v8f o1 = {};                 // ctx^T rows d=16..31
  float m = -INFINITY, lsum = 0.0f;

  for (int kc = 0; kc < LK / 32; ++kc) {
    const v16h* Kt = reinterpret_cast<const v16h*>(Kp) + (size_t)(bh * 128 + kc * 2) * 64;
    v16h ka00 = Kt[lane];        // k-tile 0, dims 0..31 (A operand)
    v16h ka01 = Kt[32 + lane];   // k-tile 0, dims 32..63
    v16h ka10 = Kt[64 + lane];   // k-tile 1, dims 0..31
    v16h ka11 = Kt[96 + lane];   // k-tile 1, dims 32..63

    v8f c0 = {}; v8f c1 = {};
    c0 = wmma16x16x32(ka00, qb0, c0);
    c0 = wmma16x16x32(ka01, qb1, c0);    // S^T rows k=kc*32+0..15
    c1 = wmma16x16x32(ka10, qb0, c1);
    c1 = wmma16x16x32(ka11, qb1, c1);    // S^T rows k=kc*32+16..31

    int kb = kc * 32 + khalf * 8;        // this lane's k rows: kb..+7, kb+16..+23
    float4 G0 = *reinterpret_cast<const float4*>(kb_row + kb);
    float4 G1 = *reinterpret_cast<const float4*>(kb_row + kb + 4);
    float4 G2 = *reinterpret_cast<const float4*>(kb_row + kb + 16);
    float4 G3 = *reinterpret_cast<const float4*>(kb_row + kb + 20);

    v16h pb = flash_update(c0, c1, lp_row, kb, G0, G1, G2, G3, m, lsum, o0, o1);

    const v16h* Vt = reinterpret_cast<const v16h*>(Vp) + (size_t)(bh * 64 + kc) * 64;
    v16h va0 = Vt[lane];         // V^T rows d=0..15, K = this 32-k chunk
    v16h va1 = Vt[32 + lane];    // V^T rows d=16..31
    o0 = wmma16x16x32(va0, pb, o0);
    o1 = wmma16x16x32(va1, pb, o1);
  }

  float inv = 1.0f / lsum;
  float* crow = ctx + (size_t)(b * LQ + q) * HIDN + h * DD + khalf * 8;
  *reinterpret_cast<float4*>(crow)      = make_float4(o0[0]*inv, o0[1]*inv, o0[2]*inv, o0[3]*inv);
  *reinterpret_cast<float4*>(crow + 4)  = make_float4(o0[4]*inv, o0[5]*inv, o0[6]*inv, o0[7]*inv);
  *reinterpret_cast<float4*>(crow + 16) = make_float4(o1[0]*inv, o1[1]*inv, o1[2]*inv, o1[3]*inv);
  *reinterpret_cast<float4*>(crow + 20) = make_float4(o1[4]*inv, o1[5]*inv, o1[6]*inv, o1[7]*inv);
}

// ---------------------------------------------------------------------------
// out = ctx @ Wo^T + bo   (4096x256 x 256x256), one wave per 16x16 tile,
// on-the-fly f32->f16 operand assembly, K=256 in 8 WMMA steps.
// ---------------------------------------------------------------------------
__global__ void outproj_kernel(const float* __restrict__ ctx,
                               const float* __restrict__ Wo,
                               const float* __restrict__ bo,
                               float* __restrict__ out) {
  int lane = threadIdx.x & 31;
  int wave = threadIdx.x >> 5;
  int tile = blockIdx.x * 4 + wave;     // exactly 4096 tiles
  int ct = tile & 15, rt = tile >> 4;
  int khalf = lane >> 4, nl = lane & 15;
  int row = rt * 16 + nl;               // A-operand row (m)
  int col = ct * 16 + nl;               // B-operand col (n)

  v8f acc = {};
#pragma unroll
  for (int kc = 0; kc < HIDN / 32; ++kc) {
    int kb = kc * 32 + khalf * 8;
    const float* ar = ctx + (size_t)row * HIDN + kb;
    const float* br = Wo  + (size_t)col * HIDN + kb;   // B[k][col] = Wo[col][k]
    float4 A0 = *reinterpret_cast<const float4*>(ar);
    float4 A1 = *reinterpret_cast<const float4*>(ar + 4);
    float4 A2 = *reinterpret_cast<const float4*>(ar + 16);
    float4 A3 = *reinterpret_cast<const float4*>(ar + 20);
    float4 B0 = *reinterpret_cast<const float4*>(br);
    float4 B1 = *reinterpret_cast<const float4*>(br + 4);
    float4 B2 = *reinterpret_cast<const float4*>(br + 16);
    float4 B3 = *reinterpret_cast<const float4*>(br + 20);
    v16h a  = cvt16(A0, A1, A2, A3);
    v16h bb = cvt16(B0, B1, B2, B3);
    acc = wmma16x16x32(a, bb, acc);
  }

  float bias = bo[col];
#pragma unroll
  for (int r = 0; r < 8; ++r) {
    int orow = rt * 16 + r + khalf * 8;
    out[(size_t)orow * HIDN + col] = acc[r] + bias;
  }
}

// ---------------------------------------------------------------------------
// launch
// ---------------------------------------------------------------------------
extern "C" void kernel_launch(void* const* d_in, const int* in_sizes, int n_in,
                              void* d_out, int out_size, void* d_ws, size_t ws_size,
                              hipStream_t stream) {
  (void)in_sizes; (void)n_in; (void)out_size; (void)ws_size;
  const float* q_fp  = (const float*)d_in[0];
  const float* v_ret = (const float*)d_in[1];
  const float* pi    = (const float*)d_in[2];
  const float* Wq_s  = (const float*)d_in[3];
  const float* bq_s  = (const float*)d_in[4];
  const float* Wk_s  = (const float*)d_in[5];
  const float* bk_s  = (const float*)d_in[6];
  const float* Wq_w  = (const float*)d_in[7];
  const float* bq_w  = (const float*)d_in[8];
  const float* Wk_w  = (const float*)d_in[9];
  const float* bk_w  = (const float*)d_in[10];
  const float* Wv    = (const float*)d_in[11];
  const float* bv    = (const float*)d_in[12];
  const float* Wo    = (const float*)d_in[13];
  const float* bo    = (const float*)d_in[14];
  const float* wb    = (const float*)d_in[15];
  float* out = (float*)d_out;

  char* ws = (char*)d_ws;
  float*    logpi = (float*)(ws);                  // 33,554,432 B  (log2 pi)
  _Float16* Qp    = (_Float16*)(ws + 33554432);    //  4,194,304 B
  _Float16* Kp    = (_Float16*)(ws + 37748736);    //  8,388,608 B
  _Float16* Vp    = (_Float16*)(ws + 46137344);    //  4,194,304 B
  float*    kbias = (float*)(ws + 50331648);       //    262,144 B
  float*    ctx   = (float*)(ws + 50593792);       //  4,194,304 B

  hipLaunchKernelGGL(logpi_kernel,  dim3(32768), dim3(256), 0, stream, pi, logpi);
  hipLaunchKernelGGL(pack_q_kernel, dim3(512),   dim3(256), 0, stream,
                     q_fp, Wq_s, bq_s, Wq_w, bq_w, wb, Qp);
  hipLaunchKernelGGL(pack_k_kernel, dim3(1024),  dim3(256), 0, stream,
                     v_ret, Wk_s, bk_s, Wk_w, bk_w, Kp);
  hipLaunchKernelGGL(pack_v_kernel, dim3(512),   dim3(256), 0, stream,
                     v_ret, Wv, bv, Vp);
  hipLaunchKernelGGL(kbias_kernel,  dim3(256),   dim3(256), 0, stream,
                     v_ret, Wk_w, bk_w, wb, kbias);
  hipLaunchKernelGGL(flash_kernel,  dim3(512),   dim3(128), 0, stream,
                     Qp, Kp, Vp, kbias, logpi, ctx);
  hipLaunchKernelGGL(outproj_kernel, dim3(1024), dim3(128), 0, stream,
                     ctx, Wo, bo, out);
}
